// LinearAttention_69355131896470
// MI455X (gfx1250) — matrix operand order repaired and checked by reading, not compiled, not run on hardware
//
#include <hip/hip_runtime.h>

// ---------------------------------------------------------------------------
// LinearAttention on MI455X (gfx1250, wave32, WMMA + async LDS staging).
//   qkv = X @ Wqkv ; K' = softmax_n(K) ; ctx = K'^T V ; M = ctx @ Wout
//   out = Q @ M + b          (fuses ctx@Wout before the big n-GEMM)
// GEMM core: block = 8 waves = 128(M) x 128(N) tile, wave = 32x64,
// B panel staged in double-buffered LDS (async global->LDS when available),
// A fed per-lane straight from global (K-contiguous), f32 accumulation.
// ---------------------------------------------------------------------------

typedef __bf16 bf16;
typedef bf16  v16bf __attribute__((ext_vector_type(16)));
typedef bf16  v8bf  __attribute__((ext_vector_type(8)));
typedef bf16  v4bf  __attribute__((ext_vector_type(4)));
typedef float v8f   __attribute__((ext_vector_type(8)));
typedef int   v4i   __attribute__((ext_vector_type(4)));

#define NB   16      // batch
#define NSP  4096    // spatial n = 64*64
#define NC   256     // channels
#define NC3  768     // 3*C
#define KC   64      // K chunk staged per LDS buffer
#define LROW 72      // padded LDS row stride (144B: 16B-aligned, conflict-free)

#if __has_builtin(__builtin_amdgcn_global_load_async_to_lds_b128) && \
    __has_builtin(__builtin_amdgcn_s_wait_asynccnt)
#define ASYNC_LDS 1
#else
#define ASYNC_LDS 0
#endif

__device__ __forceinline__ v8f wmma_bf16(v16bf a, v16bf b, v8f c) {
    return __builtin_amdgcn_wmma_f32_16x16x32_bf16(false, a, false, b, (short)0, c, false, false);
}

// A tile (16 rows x 32 K, row-major, lda=K): lanes<16 take K {0..7,16..23} of
// row (lane&15), lanes>=16 take K {8..15,24..31}.  Two contiguous 16B loads.
__device__ __forceinline__ v16bf load_a_tile(const bf16* __restrict__ A, int lda, int lane) {
    const bf16* p = A + (size_t)(lane & 15) * lda + ((lane >> 4) << 3);
    v8bf lo = *(const v8bf*)(p);
    v8bf hi = *(const v8bf*)(p + 16);
    return __builtin_shufflevector(lo, hi, 0,1,2,3,4,5,6,7,8,9,10,11,12,13,14,15);
}

// ---------------------------------------------------------------------------
// Block GEMM core: C[128x128] += A[128xK] * B[N][K]-layout.
//   A: block origin (row m_blk, k=0), row-major, lda = K stride.
//   B: block origin (row n_blk, k=0), [N][K] row-major, ldb = K stride.
// Result in acc[2][4] per wave (wave tile 32x64).
// ---------------------------------------------------------------------------
__device__ __forceinline__ void gemm_core(const bf16* __restrict__ A, int lda,
                                          const bf16* __restrict__ B, int ldb,
                                          int K, v8f acc[2][4]) {
    __shared__ bf16 lds_b[2][128][LROW];
    const int t    = threadIdx.x;
    const int lane = t & 31;
    const int wm   = (t >> 5) & 3;   // wave M position (4 x 32 rows)
    const int wn   = t >> 7;         // wave N position (2 x 64 cols)
    const int srow = t >> 1;         // staging: row per thread pair
    const int scol = (t & 1) * 32;   // staging: 32-element half-row

    auto stage = [&](int kc, int bb) {
        const bf16* src = B + (size_t)srow * ldb + kc * KC + scol;
        bf16*       dst = &lds_b[bb][srow][scol];
#if ASYNC_LDS
#pragma unroll
        for (int s = 0; s < 4; ++s)
            __builtin_amdgcn_global_load_async_to_lds_b128(
                (v4i*)(src + s * 8), (v4i*)(dst + s * 8), 0, 0);
#else
#pragma unroll
        for (int s = 0; s < 4; ++s)
            *(v8bf*)(dst + s * 8) = *(const v8bf*)(src + s * 8);
#endif
    };

    const int nkc = K / KC;
    stage(0, 0);
    for (int kc = 0; kc < nkc; ++kc) {
        const int bb = kc & 1;
#if ASYNC_LDS
        __builtin_amdgcn_s_wait_asynccnt(0);
#endif
        __syncthreads();                       // lds_b[bb] ready for all waves
        if (kc + 1 < nkc) stage(kc + 1, bb ^ 1);
#pragma unroll
        for (int kk = 0; kk < KC; kk += 32) {
            v16bf a0 = load_a_tile(A + (size_t)(wm * 32     ) * lda + kc * KC + kk, lda, lane);
            v16bf a1 = load_a_tile(A + (size_t)(wm * 32 + 16) * lda + kc * KC + kk, lda, lane);
#pragma unroll
            for (int j = 0; j < 4; ++j) {
                const bf16* bp = &lds_b[bb][wn * 64 + j * 16 + (lane & 15)][kk + ((lane >> 4) << 3)];
                v8bf blo = *(const v8bf*)(bp);
                v8bf bhi = *(const v8bf*)(bp + 16);
                v16bf bv = __builtin_shufflevector(blo, bhi,
                                                   0,1,2,3,4,5,6,7,8,9,10,11,12,13,14,15);
                acc[0][j] = wmma_bf16(a0, bv, acc[0][j]);
                acc[1][j] = wmma_bf16(a1, bv, acc[1][j]);
            }
        }
        __syncthreads();                       // reads of lds_b[bb] done
    }
}

// ---- conversion kernels ----------------------------------------------------
__global__ void cvt_f32_to_bf16_x4(const float* __restrict__ in, bf16* __restrict__ out) {
    size_t i = ((size_t)blockIdx.x * blockDim.x + threadIdx.x) * 4;
    float4 f = *(const float4*)(in + i);
    v4bf o = { (bf16)f.x, (bf16)f.y, (bf16)f.z, (bf16)f.w };
    *(v4bf*)(out + i) = o;
}
__global__ void cvt_transpose(const float* __restrict__ in, bf16* __restrict__ out,
                              int K, int N) {
    int idx = blockIdx.x * blockDim.x + threadIdx.x;
    if (idx >= K * N) return;
    int n = idx / K, k = idx - n * K;
    out[idx] = (bf16)in[(size_t)k * N + n];
}

// ---- GEMM 1: qkv = X @ Wqkv, split epilogue --------------------------------
__global__ void __launch_bounds__(256)
gemm_qkv(const bf16* __restrict__ xb, const bf16* __restrict__ wqT,
         bf16* __restrict__ q_ws, float* __restrict__ k_ws, bf16* __restrict__ v_ws) {
    int b = blockIdx.z, m_blk = blockIdx.x * 128, n_blk = blockIdx.y * 128;
    int lane = threadIdx.x & 31, wm = (threadIdx.x >> 5) & 3, wn = threadIdx.x >> 7;
    v8f acc[2][4] = {};
    gemm_core(xb + ((size_t)b * NSP + m_blk) * NC, NC, wqT + (size_t)n_blk * NC, NC, NC, acc);
    int col = lane & 15, rbase = (lane >> 4) << 3;
#pragma unroll
    for (int i = 0; i < 2; ++i) {
        int mbase = m_blk + wm * 32 + i * 16 + rbase;
#pragma unroll
        for (int j = 0; j < 4; ++j) {
            int n = n_blk + wn * 64 + j * 16 + col;
            if (n < 256) {                     // Q: [b][i][c]
                bf16* qp = q_ws + ((size_t)b * NSP + mbase) * NC + n;
#pragma unroll
                for (int r = 0; r < 8; ++r) qp[(size_t)r * NC] = (bf16)acc[i][j][r];
            } else if (n < 512) {              // K: [b][c][i], contiguous in i
                float* kp = k_ws + ((size_t)b * NC + (n - 256)) * NSP + mbase;
#pragma unroll
                for (int r = 0; r < 8; ++r) kp[r] = acc[i][j][r];
            } else {                           // V: [b][c][i]
                bf16* vp = v_ws + ((size_t)b * NC + (n - 512)) * NSP + mbase;
#pragma unroll
                for (int r = 0; r < 8; ++r) vp[r] = (bf16)acc[i][j][r];
            }
        }
    }
}

// ---- softmax over spatial axis: one block per (b, channel) row -------------
__global__ void softmax_rows(const float* __restrict__ k_ws, bf16* __restrict__ e_ws) {
    int row = blockIdx.x;
    const float* kr = k_ws + (size_t)row * NSP;
    bf16*        er = e_ws + (size_t)row * NSP;
    int t = threadIdx.x;
    float vals[16], m = -1e30f;
#pragma unroll
    for (int s = 0; s < 16; ++s) { vals[s] = kr[t + s * 256]; m = fmaxf(m, vals[s]); }
    __shared__ float red[256];
    red[t] = m; __syncthreads();
    for (int o = 128; o > 0; o >>= 1) { if (t < o) red[t] = fmaxf(red[t], red[t + o]); __syncthreads(); }
    m = red[0]; __syncthreads();
    float sum = 0.f;
#pragma unroll
    for (int s = 0; s < 16; ++s) { vals[s] = __expf(vals[s] - m); sum += vals[s]; }
    red[t] = sum; __syncthreads();
    for (int o = 128; o > 0; o >>= 1) { if (t < o) red[t] += red[t + o]; __syncthreads(); }
    float inv = 1.f / red[0];
#pragma unroll
    for (int s = 0; s < 16; ++s) er[t + s * 256] = (bf16)(vals[s] * inv);
}

// ---- GEMM 2: ctx = E^T V, split-K (4 parts of 1024) into f32 partials ------
__global__ void __launch_bounds__(256)
gemm_ctx(const bf16* __restrict__ e_ws, const bf16* __restrict__ v_ws,
         float* __restrict__ ctx_part) {
    int b = blockIdx.z >> 2, part = blockIdx.z & 3;
    int m_blk = blockIdx.x * 128, n_blk = blockIdx.y * 128;
    int lane = threadIdx.x & 31, wm = (threadIdx.x >> 5) & 3, wn = threadIdx.x >> 7;
    const bf16* A = e_ws + ((size_t)b * NC + m_blk) * NSP + part * 1024;
    const bf16* B = v_ws + ((size_t)b * NC + n_blk) * NSP + part * 1024;
    v8f acc[2][4] = {};
    gemm_core(A, NSP, B, NSP, 1024, acc);
    int col = lane & 15, rbase = (lane >> 4) << 3;
    float* cp0 = ctx_part + (((size_t)part * NB + b) * NC) * NC;
#pragma unroll
    for (int i = 0; i < 2; ++i) {
        int mbase = m_blk + wm * 32 + i * 16 + rbase;
#pragma unroll
        for (int j = 0; j < 4; ++j) {
            int n = n_blk + wn * 64 + j * 16 + col;
            float* cp = cp0 + (size_t)mbase * NC + n;
#pragma unroll
            for (int r = 0; r < 8; ++r) cp[(size_t)r * NC] = acc[i][j][r];
        }
    }
}
// deterministic 4-way reduction of split-K partials -> bf16 ctx [b][d][e]
__global__ void reduce_ctx(const float* __restrict__ ctx_part, bf16* __restrict__ ctx_bf) {
    size_t i = (size_t)blockIdx.x * blockDim.x + threadIdx.x;
    const size_t S = (size_t)NB * NC * NC;
    float s = ctx_part[i] + ctx_part[i + S] + ctx_part[i + 2 * S] + ctx_part[i + 3 * S];
    ctx_bf[i] = (bf16)s;
}

// ---- GEMM 3: M = ctx @ Wout (256^3), stored transposed [b][e_out][d] -------
__global__ void __launch_bounds__(256)
gemm_mmat(const bf16* __restrict__ ctx_bf, const bf16* __restrict__ woT,
          bf16* __restrict__ m_ws) {
    int b = blockIdx.z, m_blk = blockIdx.x * 128, n_blk = blockIdx.y * 128;
    int lane = threadIdx.x & 31, wm = (threadIdx.x >> 5) & 3, wn = threadIdx.x >> 7;
    v8f acc[2][4] = {};
    gemm_core(ctx_bf + ((size_t)b * NC + m_blk) * NC, NC, woT + (size_t)n_blk * NC, NC, NC, acc);
    int col = lane & 15, rbase = (lane >> 4) << 3;
#pragma unroll
    for (int i = 0; i < 2; ++i) {
        int mbase = m_blk + wm * 32 + i * 16 + rbase;
#pragma unroll
        for (int j = 0; j < 4; ++j) {
            int n = n_blk + wn * 64 + j * 16 + col;
            bf16* mp = m_ws + ((size_t)b * NC + n) * NC + mbase;   // [e_out][d]
#pragma unroll
            for (int r = 0; r < 8; ++r) mp[r] = (bf16)acc[i][j][r];
        }
    }
}

// ---- GEMM 4: out = Q @ M + bias (f32 output) -------------------------------
__global__ void __launch_bounds__(256)
gemm_out(const bf16* __restrict__ q_ws, const bf16* __restrict__ m_ws,
         const float* __restrict__ b_out, float* __restrict__ out) {
    int b = blockIdx.z, m_blk = blockIdx.x * 128, n_blk = blockIdx.y * 128;
    int lane = threadIdx.x & 31, wm = (threadIdx.x >> 5) & 3, wn = threadIdx.x >> 7;
    v8f acc[2][4] = {};
    gemm_core(q_ws + ((size_t)b * NSP + m_blk) * NC, NC,
              m_ws + ((size_t)b * NC + n_blk) * NC, NC, NC, acc);
    int col = lane & 15, rbase = (lane >> 4) << 3;
#pragma unroll
    for (int i = 0; i < 2; ++i) {
        int mbase = m_blk + wm * 32 + i * 16 + rbase;
#pragma unroll
        for (int j = 0; j < 4; ++j) {
            int n = n_blk + wn * 64 + j * 16 + col;
            float bias = b_out[n];
            float* op = out + ((size_t)b * NSP + mbase) * NC + n;
#pragma unroll
            for (int r = 0; r < 8; ++r) op[(size_t)r * NC] = acc[i][j][r] + bias;
        }
    }
}

// ---------------------------------------------------------------------------
extern "C" void kernel_launch(void* const* d_in, const int* in_sizes, int n_in,
                              void* d_out, int out_size, void* d_ws, size_t ws_size,
                              hipStream_t stream) {
    const float* x    = (const float*)d_in[0];
    const float* Wqkv = (const float*)d_in[1];
    const float* Wout = (const float*)d_in[2];
    const float* bout = (const float*)d_in[3];
    float*       out  = (float*)d_out;

    char* ws = (char*)d_ws;
    size_t off = 0;
    auto alloc = [&](size_t bytes) { void* p = ws + off; off = (off + bytes + 255) & ~(size_t)255; return p; };
    bf16*  x_bf     = (bf16*) alloc((size_t)NB * NSP * NC * 2);       // 32 MiB
    bf16*  wqT      = (bf16*) alloc((size_t)NC3 * NC * 2);
    bf16*  woT      = (bf16*) alloc((size_t)NC * NC * 2);
    bf16*  q_ws     = (bf16*) alloc((size_t)NB * NSP * NC * 2);       // 32 MiB
    float* k_ws     = (float*)alloc((size_t)NB * NC * NSP * 4);       // 64 MiB
    bf16*  v_ws     = (bf16*) alloc((size_t)NB * NC * NSP * 2);       // 32 MiB
    bf16*  e_ws     = (bf16*) alloc((size_t)NB * NC * NSP * 2);       // 32 MiB
    float* ctx_part = (float*)alloc((size_t)4 * NB * NC * NC * 4);    // 16 MiB
    bf16*  ctx_bf   = (bf16*) alloc((size_t)NB * NC * NC * 2);
    bf16*  m_ws     = (bf16*) alloc((size_t)NB * NC * NC * 2);
    (void)ws_size; (void)in_sizes; (void)n_in; (void)out_size;

    cvt_f32_to_bf16_x4<<<(NB * NSP * NC / 4) / 256, 256, 0, stream>>>(x, x_bf);
    cvt_transpose<<<(NC * NC3 + 255) / 256, 256, 0, stream>>>(Wqkv, wqT, NC, NC3);
    cvt_transpose<<<(NC * NC + 255) / 256, 256, 0, stream>>>(Wout, woT, NC, NC);

    gemm_qkv<<<dim3(NSP / 128, NC3 / 128, NB), 256, 0, stream>>>(x_bf, wqT, q_ws, k_ws, v_ws);
    softmax_rows<<<NB * NC, 256, 0, stream>>>(k_ws, e_ws);
    gemm_ctx<<<dim3(NC / 128, NC / 128, NB * 4), 256, 0, stream>>>(e_ws, v_ws, ctx_part);
    reduce_ctx<<<(NB * NC * NC) / 256, 256, 0, stream>>>(ctx_part, ctx_bf);
    gemm_mmat<<<dim3(NC / 128, NC / 128, NB), 256, 0, stream>>>(ctx_bf, woT, m_ws);
    gemm_out<<<dim3(NSP / 128, NC / 128, NB), 256, 0, stream>>>(q_ws, m_ws, bout, out);
}
